// RWKV6LucidTimeMix_69818988364376
// MI455X (gfx1250) — compile-verified, matrix operand order checked by reading, not run on hardware
//
#include <hip/hip_runtime.h>
#include <hip/hip_bf16.h>

// RWKV6 Lucid time-mix for MI455X (gfx1250, wave32, WMMA, async-to-LDS).
// Pipeline: convert weights -> mix -> 4x WMMA GEMM (r,k,v,g) -> chunk scan
// (LDS-resident, WMMA 64x64 matmuls + Gauss-Jordan solve) -> groupnorm*gate
// -> WMMA GEMM (W_o) ; plus x[:,-1] copy.

#define B_ 4
#define T_ 2048
#define C_ 1024
#define H_ 16
#define KK_ 64
#define CHUNK_ 64
#define NCHUNK_ 32
#define M_ (B_*T_)

typedef __attribute__((ext_vector_type(16))) __bf16 bf16x16;
typedef __attribute__((ext_vector_type(8)))  float  floatx8;
typedef __attribute__((ext_vector_type(4)))  int    intx4;

// ---- CDNA5 async global->LDS path (ASYNCcnt), guarded so either toolchain builds.
#if defined(__gfx1250__) && __has_builtin(__builtin_amdgcn_global_load_async_to_lds_b128) && __has_builtin(__builtin_amdgcn_s_wait_asynccnt)
#define HAVE_ASYNC_LDS 1
#else
#define HAVE_ASYNC_LDS 0
#endif

static __device__ __forceinline__ void cp_b128_to_lds(void* ldsDst, const void* gSrc) {
#if HAVE_ASYNC_LDS
  // signature (from hipcc diagnostic): (int4 addrspace(1)*, int4 addrspace(3)*, imm, imm)
  __builtin_amdgcn_global_load_async_to_lds_b128(
      (__attribute__((address_space(1))) intx4*)gSrc,
      (__attribute__((address_space(3))) intx4*)ldsDst, 0, 0);
#else
  *(uint4*)ldsDst = *(const uint4*)gSrc;
#endif
}
static __device__ __forceinline__ void async_wait0() {
#if HAVE_ASYNC_LDS
  __builtin_amdgcn_s_wait_asynccnt(0);
#endif
}

static __device__ __forceinline__ unsigned short f2bf(float f) {
  union { float f; unsigned u; } c; c.f = f;
  unsigned r = c.u + 0x7FFFu + ((c.u >> 16) & 1u);   // round-to-nearest-even
  return (unsigned short)(r >> 16);
}
static __device__ __forceinline__ float clip30(float x){ return fminf(fmaxf(x,-30.f),30.f); }
static __device__ __forceinline__ float clip20(float x){ return fminf(fmaxf(x,-20.f),20.f); }

// 16-bit A-matrix 16x32 fragment per ISA 7.12.2:
// lane<16: M=lane, K in {0..7, 16..23}; lane>=16: M=lane-16, K in {8..15, 24..31}.
// B fragment loaded with the mirrored layout (N<->M) from a [N][K] LDS tile.
static __device__ __forceinline__ bf16x16
load_frag(const unsigned short* lds, int rowBase, int stride, int kBase, int lane)
{
  int r  = rowBase + (lane & 15);
  int kh = (lane >> 4) << 3;                       // 0 or 8
  const unsigned short* p = lds + r*stride + kBase + kh;
  union { unsigned u[8]; bf16x16 v; } f;
  f.u[0] = *(const unsigned*)(p + 0);
  f.u[1] = *(const unsigned*)(p + 2);
  f.u[2] = *(const unsigned*)(p + 4);
  f.u[3] = *(const unsigned*)(p + 6);
  f.u[4] = *(const unsigned*)(p + 16);
  f.u[5] = *(const unsigned*)(p + 18);
  f.u[6] = *(const unsigned*)(p + 20);
  f.u[7] = *(const unsigned*)(p + 22);
  return f.v;
}

// Out[m][n] = sum_k aB[m][k] * bB[n][k]  over 64x64x64, 16 tiles / 8 waves.
static __device__ __forceinline__ void
mm64_lds(const unsigned short* aB, const unsigned short* bB,
         float* Out, int wave, int lane, bool add)
{
  for (int i = 0; i < 2; ++i) {
    int tile = wave*2 + i;
    int tm = (tile >> 2) << 4;
    int tn = (tile & 3)  << 4;
    floatx8 acc = {};
    for (int kk = 0; kk < 64; kk += 32) {
      bf16x16 a = load_frag(aB, tm, 64, kk, lane);
      bf16x16 b = load_frag(bB, tn, 64, kk, lane);
      acc = __builtin_amdgcn_wmma_f32_16x16x32_bf16(false, a, false, b,
                                                    (short)0, acc, false, false);
    }
    int col = tn + (lane & 15);
    int r0  = tm + ((lane >> 4) << 3);
    for (int r = 0; r < 8; ++r) {
      if (add) Out[(r0 + r)*64 + col] += acc[r];
      else     Out[(r0 + r)*64 + col]  = acc[r];
    }
  }
}

// ---------------------------------------------------------------- weights->bf16
__global__ __launch_bounds__(256) void convert_w_kernel(
  const float* __restrict__ a, const float* __restrict__ b, const float* __restrict__ c,
  const float* __restrict__ d, const float* __restrict__ e, unsigned short* __restrict__ out)
{
  size_t i = (size_t)blockIdx.x*256 + threadIdx.x;         // 5 * 2^20 total
  const float* srcs[5] = {a, b, c, d, e};
  size_t which = i >> 20, off = i & 1048575u;
  out[i] = f2bf(srcs[which][off]);
}

// ---------------------------------------------------------------- token mixing
__global__ __launch_bounds__(256) void mix_kernel(
  const float* __restrict__ x, const float* __restrict__ shift,
  const float* __restrict__ tmx, const float* __restrict__ tmw, const float* __restrict__ tmk,
  const float* __restrict__ tmv, const float* __restrict__ tmr, const float* __restrict__ tmg,
  const float* __restrict__ w1, const float* __restrict__ w2,
  const float* __restrict__ tdecay, const float* __restrict__ dw1, const float* __restrict__ dw2,
  unsigned short* __restrict__ xr_o, unsigned short* __restrict__ xk_o,
  unsigned short* __restrict__ xv_o, unsigned short* __restrict__ xg_o,
  float* __restrict__ wlog_o)
{
  __shared__ float xs[C_], dxs[C_], mxs[C_], xxxs[160], mbuf[5*C_], wws[64];
  const int tid = threadIdx.x;
  const int token = blockIdx.x;
  const int t = token & (T_-1);
  const int b = token >> 11;
  const size_t base = (size_t)token * C_;

  for (int i = tid; i < C_; i += 256) {
    float xv = x[base + i];
    float xp = (t == 0) ? shift[(size_t)b*C_ + i] : x[base - C_ + i];
    float dx = xp - xv;
    xs[i] = xv; dxs[i] = dx;
    mxs[i] = xv + dx * tmx[i];
  }
  __syncthreads();
  if (tid < 160) {
    float s = 0.f;
    for (int c = 0; c < C_; ++c) s += mxs[c] * w1[c*160 + tid];
    xxxs[tid] = tanhf(s);
  }
  __syncthreads();
  for (int idx = tid; idx < 5*C_; idx += 256) {
    int f = idx >> 10, c = idx & (C_-1);
    float s = 0.f;
    for (int d = 0; d < 32; ++d) s += xxxs[f*32 + d] * w2[(size_t)(f*32 + d)*C_ + c];
    mbuf[idx] = s;
  }
  __syncthreads();
  for (int i = tid; i < C_; i += 256) {
    float xv = xs[i], dx = dxs[i];
    float xw = xv + dx * (tmw[i] + mbuf[i]);
    xk_o[base + i] = f2bf(xv + dx*(tmk[i] + mbuf[C_   + i]));
    xv_o[base + i] = f2bf(xv + dx*(tmv[i] + mbuf[2*C_ + i]));
    xr_o[base + i] = f2bf(xv + dx*(tmr[i] + mbuf[3*C_ + i]));
    xg_o[base + i] = f2bf(xv + dx*(tmg[i] + mbuf[4*C_ + i]));
    mxs[i] = xw;                                   // reuse for decay branch
  }
  __syncthreads();
  if (tid < 64) {
    float s = 0.f;
    for (int c = 0; c < C_; ++c) s += mxs[c] * dw1[c*64 + tid];
    wws[tid] = tanhf(s);
  }
  __syncthreads();
  for (int i = tid; i < C_; i += 256) {
    float s = tdecay[i];
    for (int d = 0; d < 64; ++d) s += wws[d] * dw2[(size_t)d*C_ + i];
    wlog_o[base + i] = -expf(s);                   // w_h = -exp(w)
  }
}

// -------- WMMA GEMM 128x128 tile, double-buffered async LDS staging.
// Out[m][n] = sum_k A[m,k]*W[n,k]  (M=8192, N=K=1024).
__global__ __launch_bounds__(256) void gemm_kernel(
  const unsigned short* __restrict__ A, const unsigned short* __restrict__ W,
  float* __restrict__ Out, int act)
{
  __shared__ unsigned short As[2][128*32];
  __shared__ unsigned short Bs[2][128*32];
  const int tid = threadIdx.x, lane = tid & 31, wave = tid >> 5;
  const int bm = blockIdx.x * 128, bn = blockIdx.y * 128;
  const int wm = (wave & 3) * 32, wn = (wave >> 2) * 64;

  floatx8 z = {};
  floatx8 acc[2][4];
#pragma unroll
  for (int i = 0; i < 2; ++i)
#pragma unroll
    for (int j = 0; j < 4; ++j) acc[i][j] = z;

  auto stage = [&](int buf, int k0) {
#pragma unroll
    for (int i = 0; i < 2; ++i) {
      int idx = tid*2 + i;                         // 0..511 : 128 rows x 4 segs
      int r = idx >> 2, seg = idx & 3;
      cp_b128_to_lds(&As[buf][r*32 + seg*8], A + (size_t)(bm + r)*C_ + k0 + seg*8);
      cp_b128_to_lds(&Bs[buf][r*32 + seg*8], W + (size_t)(bn + r)*C_ + k0 + seg*8);
    }
  };

  stage(0, 0);
  for (int k0 = 0; k0 < C_; k0 += 32) {
    const int cur = (k0 >> 5) & 1;
    async_wait0();
    __syncthreads();
    if (k0 + 32 < C_) {
      stage(cur ^ 1, k0 + 32);                     // overlap copy with WMMA
      __builtin_prefetch(A + (size_t)(bm + (tid >> 1))*C_ + k0 + 64, 0, 1);
    }
    bf16x16 fa0 = load_frag(As[cur], wm,    32, 0, lane);
    bf16x16 fa1 = load_frag(As[cur], wm+16, 32, 0, lane);
#pragma unroll
    for (int j = 0; j < 4; ++j) {
      bf16x16 fb = load_frag(Bs[cur], wn + j*16, 32, 0, lane);
      acc[0][j] = __builtin_amdgcn_wmma_f32_16x16x32_bf16(false, fa0, false, fb, (short)0, acc[0][j], false, false);
      acc[1][j] = __builtin_amdgcn_wmma_f32_16x16x32_bf16(false, fa1, false, fb, (short)0, acc[1][j], false, false);
    }
    __syncthreads();
  }

  const int col = lane & 15, rOff = (lane >> 4) * 8;
#pragma unroll
  for (int fm = 0; fm < 2; ++fm)
#pragma unroll
    for (int j = 0; j < 4; ++j) {
      size_t rb = (size_t)(bm + wm + fm*16 + rOff);
      int    cb = bn + wn + j*16 + col;
#pragma unroll
      for (int r = 0; r < 8; ++r) {
        float v = acc[fm][j][r];
        if (act) v = v / (1.f + expf(-v));         // silu for the g branch
        Out[(rb + r)*C_ + cb] = v;
      }
    }
}

// ---------------------------------------------------------------- chunk scan
__global__ __launch_bounds__(256) void chunk_kernel(
  const float* __restrict__ r_g, const float* __restrict__ k_g, const float* __restrict__ v_g,
  const float* __restrict__ wl_g, const float* __restrict__ state_in,
  const float* __restrict__ faaaa, const float* __restrict__ lucid,
  float* __restrict__ y_g, float* __restrict__ state_out)
{
  extern __shared__ char smemRaw[];
  float* state = (float*)smemRaw;                  // [64k][64v]
  float* r_s  = state + 4096;
  float* k_s  = r_s  + 4096;
  float* v_s  = k_s  + 4096;
  float* wl   = v_s  + 4096;
  float* wc   = wl   + 4096;
  float* Am   = wc   + 4096;
  float* Pm   = Am   + 4096;
  float* PV   = Pm   + 4096;
  float* wsum = PV   + 4096;   // [64]
  float* diag = wsum + 64;     // [64]
  float* rn   = diag + 64;     // [64]
  float* u_s  = rn   + 64;     // [64]
  float* sc   = u_s  + 64;     // [0]=temp, [1]=pivinv
  unsigned short* aB = (unsigned short*)(sc + 8);  // [64][64] bf16
  unsigned short* bB = aB + 4096;

  const int tid = threadIdx.x, lane = tid & 31, wave = tid >> 5;
  const int bh = blockIdx.x, b = bh >> 4, h = bh & 15;

  for (int q = tid; q < 1024; q += 256)
    cp_b128_to_lds(&state[q*4], &state_in[(size_t)bh*4096 + q*4]);
  if (tid < 64) u_s[tid] = faaaa[h*64 + tid];
  if (tid == 0) sc[0] = log1pf(expf(lucid[h]));    // softplus(temperature)
  async_wait0();
  __syncthreads();
  const float temp = sc[0];

  for (int c = 0; c < NCHUNK_; ++c) {
    const size_t rowBase = ((size_t)b*T_ + (size_t)c*CHUNK_)*C_ + (size_t)h*64;

    for (int q = tid; q < 1024; q += 256) {        // async 16B tile loads
      int t = q >> 4, k4 = (q & 15) << 2;
      size_t g = rowBase + (size_t)t*C_ + k4;
      int l = t*64 + k4;
      cp_b128_to_lds(&r_s[l], &r_g[g]);
      cp_b128_to_lds(&k_s[l], &k_g[g]);
      cp_b128_to_lds(&v_s[l], &v_g[g]);
      cp_b128_to_lds(&wl[l],  &wl_g[g]);
    }
    async_wait0();
    __syncthreads();
    if (tid < 64) {                                // cumsum along t, per k-column
      float s = 0.f;
      for (int t = 0; t < 64; ++t) { s += wl[t*64 + tid]; wc[t*64 + tid] = s; }
      wsum[tid] = s;
    }
    __syncthreads();
    if (tid < 64) {                                // diag and 1/||k|| per row
      float s = 0.f, n2 = 0.f;
      for (int kk = 0; kk < 64; ++kk) {
        float rv = r_s[tid*64 + kk], kv = k_s[tid*64 + kk];
        s  += rv * u_s[kk] * kv;
        n2 += kv * kv;
      }
      diag[tid] = s;
      rn[tid] = 1.f / fmaxf(sqrtf(n2), 1e-12f);
    }
    __syncthreads();
    // A = (r*r_decay) @ (k*k_inv_decay)^T
    for (int idx = tid; idx < 4096; idx += 256) {
      int t = idx >> 6, kk = idx & 63;
      float off = wc[31*64 + kk];                  // w_shift at mid=32
      float wsh = t ? wc[(t-1)*64 + kk] : 0.f;
      aB[idx] = f2bf(r_s[idx] * expf(clip30(wsh - off)));
      bB[idx] = f2bf(k_s[idx] * expf(clip30(off - wc[idx])));
    }
    __syncthreads();
    mm64_lds(aB, bB, Am, wave, lane, false);
    __syncthreads();
    // tril(A,-1)+diag ; stage knorm for K_gram ; PV = v
    for (int idx = tid; idx < 4096; idx += 256) {
      int t = idx >> 6, s2 = idx & 63;
      float av = Am[idx];
      Am[idx] = (s2 < t) ? av : (s2 == t ? diag[t] : 0.f);
      unsigned short kn = f2bf(k_s[idx] * rn[t]);
      aB[idx] = kn; bB[idx] = kn;
      PV[idx] = v_s[idx];
    }
    __syncthreads();
    mm64_lds(aB, bB, Pm, wave, lane, false);       // K_gram
    __syncthreads();
    for (int idx = tid; idx < 4096; idx += 256) {
      int t = idx >> 6, s2 = idx & 63;
      Pm[idx] = ((t == s2) ? 1.f : 0.f) + expf(clip20(temp * Pm[idx]));
    }
    __syncthreads();
    // Gauss-Jordan: solve P X = v  (X ends in PV)
    for (int piv = 0; piv < 64; ++piv) {
      if (tid == 0) sc[1] = 1.f / Pm[piv*64 + piv];
      __syncthreads();
      float pinv = sc[1];
      if (tid < 128) {
        if (tid < 64) Pm[piv*64 + tid] *= pinv;
        else          PV[piv*64 + tid - 64] *= pinv;
      }
      __syncthreads();
      for (int idx = tid; idx < 8192; idx += 256) {
        int i = idx >> 7, j = idx & 127;
        if (i == piv) continue;
        float f = Pm[i*64 + piv];
        if (j < 64) { if (j != piv) Pm[i*64 + j] -= f * Pm[piv*64 + j]; }
        else        PV[i*64 + j - 64] -= f * PV[piv*64 + j - 64];
      }
      __syncthreads();
    }
    // y = A @ X
    for (int idx = tid; idx < 4096; idx += 256) {
      int n = idx >> 6, s2 = idx & 63;
      aB[idx] = f2bf(Am[idx]);
      bB[idx] = f2bf(PV[s2*64 + n]);               // bB[n][s] = X[s][n]
    }
    __syncthreads();
    mm64_lds(aB, bB, Pm, wave, lane, false);       // Pm := Y
    __syncthreads();
    // y += (r * w_intra) @ state
    for (int idx = tid; idx < 4096; idx += 256) {
      int n = idx >> 6, kk = idx & 63;
      aB[idx] = f2bf(r_s[idx] * expf(clip30(wc[idx] - wl[idx])));
      bB[idx] = f2bf(state[kk*64 + n]);            // bB[n][k] = state[k][n]
    }
    __syncthreads();
    mm64_lds(aB, bB, Pm, wave, lane, true);
    __syncthreads();
    // write y (vectorized) ; stage state-update operands: (k*w_inter)^T @ v
    for (int q = tid; q < 1024; q += 256) {
      int t = q >> 4, k4 = (q & 15) << 2;
      *(float4*)&y_g[rowBase + (size_t)t*C_ + k4] = *(float4*)&Pm[t*64 + k4];
    }
    for (int idx = tid; idx < 4096; idx += 256) {
      int a_k = idx >> 6, a_t = idx & 63;          // aB[k][t]
      aB[idx] = f2bf(k_s[a_t*64 + a_k] * expf(clip30(wsum[a_k] - wc[a_t*64 + a_k])));
      bB[idx] = f2bf(v_s[a_t*64 + a_k]);           // bB[v][t] = v_s[t][v]
    }
    __syncthreads();
    mm64_lds(aB, bB, Am, wave, lane, false);
    __syncthreads();
    // state = state * exp(w_sum) (broadcast over v axis, as in reference) + kv
    for (int idx = tid; idx < 4096; idx += 256) {
      state[idx] = state[idx] * expf(wsum[idx & 63]) + Am[idx];
    }
    __syncthreads();
  }
  for (int q = tid; q < 1024; q += 256)
    *(float4*)&state_out[(size_t)bh*4096 + q*4] = *(float4*)&state[q*4];
}

// ---------------------------------------------------------------- groupnorm*gate
__global__ __launch_bounds__(256) void post_kernel(
  const float* __restrict__ y, const float* __restrict__ g,
  const float* __restrict__ gamma, const float* __restrict__ beta,
  unsigned short* __restrict__ aout)
{
  __shared__ float ys[C_];
  __shared__ float mu[H_], rs[H_];
  const int tid = threadIdx.x;
  const size_t row = blockIdx.x;
  for (int q = tid; q < C_/4; q += 256)
    *(float4*)&ys[q*4] = *(const float4*)&y[row*C_ + q*4];
  __syncthreads();
  if (tid < H_) {
    float s = 0.f, s2 = 0.f;
    for (int k = 0; k < 64; ++k) { float v = ys[tid*64 + k]; s += v; s2 += v*v; }
    float m = s * (1.f/64.f);
    float var = s2 * (1.f/64.f) - m*m;
    mu[tid] = m; rs[tid] = rsqrtf(var + 6.4e-4f);  // eps = 1e-5 * 64
  }
  __syncthreads();
  for (int i = tid; i < C_; i += 256) {
    int hh = i >> 6;
    float yn = (ys[i] - mu[hh]) * rs[hh] * gamma[i] + beta[i];
    aout[row*C_ + i] = f2bf(yn * g[row*C_ + i]);   // g already silu'd
  }
}

// ---------------------------------------------------------------- x[:, -1]
__global__ __launch_bounds__(256) void lastx_kernel(const float* __restrict__ x,
                                                    float* __restrict__ out)
{
  int i = blockIdx.x*256 + threadIdx.x;
  if (i < B_*C_) {
    int b = i >> 10, cc = i & (C_-1);
    out[i] = x[((size_t)b*T_ + (T_-1))*C_ + cc];
  }
}

// =============================================================== launcher
extern "C" void kernel_launch(void* const* d_in, const int* in_sizes, int n_in,
                              void* d_out, int out_size, void* d_ws, size_t ws_size,
                              hipStream_t stream)
{
  (void)in_sizes; (void)n_in; (void)out_size; (void)ws_size;
  const float* x      = (const float*)d_in[0];
  const float* shift  = (const float*)d_in[1];
  const float* wkv    = (const float*)d_in[2];
  const float* tmx    = (const float*)d_in[3];
  const float* tmw    = (const float*)d_in[4];
  const float* tmk    = (const float*)d_in[5];
  const float* tmv    = (const float*)d_in[6];
  const float* tmr    = (const float*)d_in[7];
  const float* tmg    = (const float*)d_in[8];
  const float* w1     = (const float*)d_in[9];
  const float* w2     = (const float*)d_in[10];
  const float* tdecay = (const float*)d_in[11];
  const float* dw1    = (const float*)d_in[12];
  const float* dw2    = (const float*)d_in[13];
  const float* faaaa  = (const float*)d_in[14];
  const float* lucid  = (const float*)d_in[15];
  const float* Wr     = (const float*)d_in[16];
  const float* Wk     = (const float*)d_in[17];
  const float* Wv     = (const float*)d_in[18];
  const float* Wg     = (const float*)d_in[19];
  const float* Wo     = (const float*)d_in[20];
  const float* gamma  = (const float*)d_in[21];
  const float* beta   = (const float*)d_in[22];

  const size_t C2 = (size_t)C_ * C_;        // 1M
  const size_t MC = (size_t)M_ * C_;        // 8M

  char* p = (char*)d_ws;
  auto take = [&](size_t bytes) -> char* {
    char* q = p; p += (bytes + 255) & ~(size_t)255; return q;
  };
  unsigned short* wbf  = (unsigned short*)take(5*C2*2);  // r,k,v,g,o bf16
  unsigned short* xr   = (unsigned short*)take(MC*2);
  unsigned short* xk   = (unsigned short*)take(MC*2);
  unsigned short* xv   = (unsigned short*)take(MC*2);
  unsigned short* xg   = (unsigned short*)take(MC*2);
  float*          rf   = (float*)take(MC*4);
  float*          kf   = (float*)take(MC*4);
  float*          vf   = (float*)take(MC*4);
  float*          gf   = (float*)take(MC*4);
  float*          wlog = (float*)take(MC*4);
  float*          yf   = (float*)take(MC*4);
  unsigned short* ao   = (unsigned short*)take(MC*2);

  float* out_main  = (float*)d_out;                     // (B,T,C)
  float* out_lastx = out_main + MC;                     // (B,C)
  float* out_state = out_lastx + (size_t)B_*C_;         // (B,H,K,K)

  convert_w_kernel<<<(5*C2)/256, 256, 0, stream>>>(Wr, Wk, Wv, Wg, Wo, wbf);

  mix_kernel<<<M_, 256, 0, stream>>>(x, shift, tmx, tmw, tmk, tmv, tmr, tmg,
                                     w1, w2, tdecay, dw1, dw2,
                                     xr, xk, xv, xg, wlog);

  dim3 gg(M_/128, C_/128);
  gemm_kernel<<<gg, 256, 0, stream>>>(xr, wbf + 0*C2, rf, 0);
  gemm_kernel<<<gg, 256, 0, stream>>>(xk, wbf + 1*C2, kf, 0);
  gemm_kernel<<<gg, 256, 0, stream>>>(xv, wbf + 2*C2, vf, 0);
  gemm_kernel<<<gg, 256, 0, stream>>>(xg, wbf + 3*C2, gf, 1);   // silu

  size_t ldsBytes = (size_t)(9*4096 + 4*64 + 8) * sizeof(float)
                  + (size_t)2*4096*sizeof(unsigned short);      // ~161 KB
  chunk_kernel<<<B_*H_, 256, ldsBytes, stream>>>(rf, kf, vf, wlog, wkv,
                                                 faaaa, lucid, yf, out_state);

  post_kernel<<<M_, 256, 0, stream>>>(yf, gf, gamma, beta, ao);

  gemm_kernel<<<gg, 256, 0, stream>>>(ao, wbf + 4*C2, out_main, 0);

  lastx_kernel<<<(B_*C_ + 255)/256, 256, 0, stream>>>(x, out_lastx);
}
